// CustomLSTM_30442728194478
// MI455X (gfx1250) — matrix operand order repaired and checked by reading, not compiled, not run on hardware
//
#include <hip/hip_runtime.h>
#include <cstdint>
#include <cstddef>

// ---------------------------------------------------------------------------
// CustomLSTM on MI455X (gfx1250, wave32, WMMA).
//   B=64, T=2048, F=256, H=512, P=256, 4H=2048
// Phase 1: x_proj GEMM (parallel, bf16 WMMA, fragment-layout output in ws)
// Phase 2: persistent recurrent kernel, 4 WGs (16 batch rows each), c-state in
//          VGPR accumulators, h exchanged through double-buffered LDS.
// ---------------------------------------------------------------------------

typedef __attribute__((ext_vector_type(16))) __bf16    v16bf;
typedef __attribute__((ext_vector_type(8)))  float     v8f;
typedef __attribute__((ext_vector_type(4)))  float     v4f;
typedef __attribute__((ext_vector_type(4)))  unsigned  v4u;

#define LSTM_B   64
#define LSTM_T   2048
#define LSTM_F   256
#define LSTM_H   512
#define LSTM_P   256
#define LSTM_G   2048   // 4*H

// workspace layout (bytes)
#define OFF_WIH   ((size_t)0)                       // 2048*256 bf16  = 1,048,576
#define OFF_WHH   ((size_t)1048576)                 // 2048*512 bf16  = 2,097,152
#define OFF_WLIN  ((size_t)3145728)                 // 256*512  bf16  =   262,144
#define OFF_BIAS  ((size_t)3407872)                 // 2048 f32       =     8,192
#define OFF_XF    ((size_t)3416064)                 // T*4*128 tiles * 32 lanes * 16B = 536,870,912

union Frag16 {            // one WMMA bf16 operand (8 VGPRs / 32B per lane)
    v4u   q[2];
    v16bf v;
    __bf16 b[16];
};

__device__ inline unsigned pack_bf16x2(float lo, float hi) {
    union { __bf16 b[2]; unsigned u; } p;
    p.b[0] = (__bf16)lo; p.b[1] = (__bf16)hi;
    return p.u;
}
__device__ inline void unpack_bf16x2(unsigned u, float& lo, float& hi) {
    union { unsigned u; __bf16 b[2]; } p; p.u = u;
    lo = (float)p.b[0]; hi = (float)p.b[1];
}
__device__ inline float sigmoidf_(float x) { return 1.0f / (1.0f + __expf(-x)); }
__device__ inline float tanhf_(float x) {
    float e = __expf(-2.0f * fabsf(x));
    float t = (1.0f - e) / (1.0f + e);
    return copysignf(t, x);
}

// --------------------------- prep: fp32 -> bf16 weights --------------------
__global__ __launch_bounds__(256)
void lstm_prep(const float* __restrict__ W_ih, const float* __restrict__ W_hh,
               const float* __restrict__ b_ih, const float* __restrict__ b_hh,
               const float* __restrict__ W_lin,
               __bf16* __restrict__ W_ih_bf, __bf16* __restrict__ W_hh_bf,
               __bf16* __restrict__ W_lin_bf, float* __restrict__ bias_sum)
{
    const int NIH = LSTM_G * LSTM_F, NHH = LSTM_G * LSTM_H, NLN = LSTM_P * LSTM_H;
    const int total = NIH + NHH + NLN + LSTM_G;
    for (int i = blockIdx.x * blockDim.x + threadIdx.x; i < total;
         i += gridDim.x * blockDim.x) {
        if (i < NIH)                 W_ih_bf[i] = (__bf16)W_ih[i];
        else if (i < NIH + NHH)      W_hh_bf[i - NIH] = (__bf16)W_hh[i - NIH];
        else if (i < NIH + NHH + NLN)W_lin_bf[i - NIH - NHH] = (__bf16)W_lin[i - NIH - NHH];
        else { int g = i - NIH - NHH - NLN; bias_sum[g] = b_ih[g] + b_hh[g]; }
    }
}

// --------------------------- phase 1: x_proj GEMM --------------------------
// grid (T, 4 groups, 2), block 256 (8 waves). Wave owns 8 N-tiles, K=F=256.
// Output: bf16-packed accumulator fragments + folded biases, lane-linear.
__global__ __launch_bounds__(256)
void lstm_xproj(const float* __restrict__ batch, const __bf16* __restrict__ W_ih_bf,
                const float* __restrict__ bias_sum, v4u* __restrict__ xf)
{
    const int t    = blockIdx.x;
    const int grp  = blockIdx.y;
    const int zb   = blockIdx.z;
    const int lane = threadIdx.x & 31;
    const int w    = threadIdx.x >> 5;
    const int n16  = lane & 15;
    const int khA  = (lane >> 4) * 8;    // A: per-lane K chunks of 8 (doc layout)
    const int khB  = (lane >> 4) * 8;    // B: per-lane K chunks of 8 (doc layout)

    v8f acc[8];
    const v8f zero8 = {0.f,0.f,0.f,0.f,0.f,0.f,0.f,0.f};
#pragma unroll
    for (int i = 0; i < 8; ++i) acc[i] = zero8;

    const int b_row = grp * 16 + n16;
    const float* arow = batch + ((size_t)b_row * LSTM_T + t) * LSTM_F;

    for (int kk = 0; kk < LSTM_F / 32; ++kk) {           // 8 K-steps of 32
        // A fragment: 16 fp32 -> bf16 (K = kk*32 + {khA..khA+7, khA+16..khA+23})
        const float* ap = arow + kk * 32 + khA;
        v4f f0 = *(const v4f*)(ap);      v4f f1 = *(const v4f*)(ap + 4);
        v4f f2 = *(const v4f*)(ap + 16); v4f f3 = *(const v4f*)(ap + 20);
        Frag16 fa;
#pragma unroll
        for (int j = 0; j < 4; ++j) {
            fa.b[j]      = (__bf16)f0[j];
            fa.b[4 + j]  = (__bf16)f1[j];
            fa.b[8 + j]  = (__bf16)f2[j];
            fa.b[12 + j] = (__bf16)f3[j];
        }
#pragma unroll
        for (int i = 0; i < 8; ++i) {
            const int tg = zb * 64 + w * 8 + i;          // N-tile 0..127
            const __bf16* brow = W_ih_bf + (size_t)(tg * 16 + n16) * LSTM_F + kk * 32 + khB;
            Frag16 fb;
            fb.q[0] = *(const v4u*)(brow);
            fb.q[1] = *(const v4u*)(brow + 16);
            acc[i] = __builtin_amdgcn_wmma_f32_16x16x32_bf16(
                         false, fa.v, false, fb.v, (short)0, acc[i], false, false);
        }
    }
#pragma unroll
    for (int i = 0; i < 8; ++i) {
        const int tg = zb * 64 + w * 8 + i;
        const float bias = bias_sum[tg * 16 + n16];
        v4u o;
#pragma unroll
        for (int d = 0; d < 4; ++d)
            o[d] = pack_bf16x2(acc[i][2 * d] + bias, acc[i][2 * d + 1] + bias);
        xf[(((size_t)t * 4 + grp) * 128 + tg) * 32 + lane] = o;
    }
}

// --------------------------- phase 2: persistent recurrence ----------------
// grid 4 (batch groups), block 512 = 16 waves. Wave w owns h/c columns
// [32w, 32w+32): tiles jt=2w,2w+1 of each gate -> i,f,g,o all in-wave,
// c stays in VGPR accumulators for all 2048 steps.
#define HPAD 520   // 512 + 8 bf16 pad -> conflict-free ds_load_b128
__global__ __launch_bounds__(512)
void lstm_step(const v4u* __restrict__ xf, const __bf16* __restrict__ W_hh_bf,
               const __bf16* __restrict__ W_lin_bf, const float* __restrict__ b_lin,
               float* __restrict__ out_h, float* __restrict__ out_c,
               float* __restrict__ out_o)
{
    __shared__ __bf16 h_lds[2][16][HPAD];   // double-buffered hidden state (bf16)
    __shared__ float  s_lds[16][260];       // pre-softmax scores [b][p]
    __shared__ float  s_mls[16];            // per-row (max + log-sum-exp)

    const int grp  = blockIdx.x;
    const int lane = threadIdx.x & 31;
    const int w    = threadIdx.x >> 5;      // 0..15
    const int n16  = lane & 15;
    const int khA  = (lane >> 4) * 8;
    const int khB  = (lane >> 4) * 8;
    const v8f zero8 = {0.f,0.f,0.f,0.f,0.f,0.f,0.f,0.f};

    // zero h(t=0)
    for (int i = threadIdx.x; i < 16 * HPAD; i += blockDim.x)
        h_lds[0][i / HPAD][i % HPAD] = (__bf16)0.0f;
    __syncthreads();

    v8f c_acc[2] = { zero8, zero8 };        // persistent cell state

    for (int t = 0; t < LSTM_T; ++t) {
        const int cur = t & 1, nxt = cur ^ 1;
        if (t + 1 < LSTM_T)                  // warm next step's x_proj tiles
            __builtin_prefetch(xf + (((size_t)(t + 1) * 4 + grp) * 128 + 2 * w) * 32 + lane, 0, 0);

        // ---- issue x_proj fragment loads EARLY (independent of LDS h) so the
        //      L2 latency is hidden under the 128-WMMA gate matmul ----
        v4u xq[4][2];
#pragma unroll
        for (int g = 0; g < 4; ++g)
#pragma unroll
            for (int r = 0; r < 2; ++r)
                xq[g][r] = xf[(((size_t)t * 4 + grp) * 128 + (g * 32 + 2 * w + r)) * 32 + lane];

        // ---- gates = h @ W_hh^T (accumulate), tiles: g*32 + (2w+r) ----
        v8f acc[4][2];
#pragma unroll
        for (int g = 0; g < 4; ++g) { acc[g][0] = zero8; acc[g][1] = zero8; }

        for (int kk = 0; kk < LSTM_H / 32; ++kk) {       // 16 K-steps
            Frag16 fa;                                    // A from LDS (shared over g,r)
            fa.q[0] = *(const v4u*)&h_lds[cur][n16][kk * 32 + khA];
            fa.q[1] = *(const v4u*)&h_lds[cur][n16][kk * 32 + khA + 16];
#pragma unroll
            for (int g = 0; g < 4; ++g)
#pragma unroll
                for (int r = 0; r < 2; ++r) {
                    const int tg = g * 32 + 2 * w + r;   // global gate tile 0..127
                    const __bf16* brow = W_hh_bf + (size_t)(tg * 16 + n16) * LSTM_H + kk * 32 + khB;
                    Frag16 fb;
                    fb.q[0] = *(const v4u*)(brow);
                    fb.q[1] = *(const v4u*)(brow + 16);
                    acc[g][r] = __builtin_amdgcn_wmma_f32_16x16x32_bf16(
                                    false, fa.v, false, fb.v, (short)0, acc[g][r], false, false);
                }
        }

        // ---- + x_proj, activations, c/h update, store h,c ----
#pragma unroll
        for (int r = 0; r < 2; ++r) {
            const int jt = 2 * w + r;                    // h-column tile 0..31
#pragma unroll
            for (int g = 0; g < 4; ++g) {
#pragma unroll
                for (int d = 0; d < 4; ++d) {
                    float lo, hi; unpack_bf16x2(xq[g][r][d], lo, hi);
                    acc[g][r][2 * d]     += lo;
                    acc[g][r][2 * d + 1] += hi;
                }
            }
            const int hcol = jt * 16 + n16;
#pragma unroll
            for (int e = 0; e < 8; ++e) {
                const float ig = sigmoidf_(acc[0][r][e]);
                const float fg = sigmoidf_(acc[1][r][e]);
                const float gg = tanhf_  (acc[2][r][e]);
                const float og = sigmoidf_(acc[3][r][e]);
                const float c  = fg * c_acc[r][e] + ig * gg;
                const float h  = og * tanhf_(c);
                c_acc[r][e] = c;
                const int mrow = e + 8 * (lane >> 4);    // accumulator row map
                h_lds[nxt][mrow][hcol] = (__bf16)h;
                const size_t base = ((size_t)t * LSTM_B + grp * 16 + mrow) * LSTM_H + hcol;
                out_h[base] = h;
                out_c[base] = c;
            }
        }
        __syncthreads();                                  // h(t) visible

        // ---- scores = h @ W_lin^T + b_lin ; wave w -> P-tile w ----
        {
            v8f sa = zero8;
            for (int kk = 0; kk < LSTM_H / 32; ++kk) {
                Frag16 fa;
                fa.q[0] = *(const v4u*)&h_lds[nxt][n16][kk * 32 + khA];
                fa.q[1] = *(const v4u*)&h_lds[nxt][n16][kk * 32 + khA + 16];
                const __bf16* brow = W_lin_bf + (size_t)(w * 16 + n16) * LSTM_H + kk * 32 + khB;
                Frag16 fb;
                fb.q[0] = *(const v4u*)(brow);
                fb.q[1] = *(const v4u*)(brow + 16);
                sa = __builtin_amdgcn_wmma_f32_16x16x32_bf16(
                         false, fa.v, false, fb.v, (short)0, sa, false, false);
            }
            const float bl = b_lin[w * 16 + n16];
#pragma unroll
            for (int e = 0; e < 8; ++e)
                s_lds[e + 8 * (lane >> 4)][w * 16 + n16] = sa[e] + bl;
        }
        __syncthreads();                                  // scores visible

        // ---- log_softmax reduction over P (wave w handles batch row w) ----
        {
            float sv[8]; float m = -3.4e38f;
#pragma unroll
            for (int k = 0; k < 8; ++k) { sv[k] = s_lds[w][lane + 32 * k]; m = fmaxf(m, sv[k]); }
#pragma unroll
            for (int off = 16; off >= 1; off >>= 1) m = fmaxf(m, __shfl_xor(m, off));
            float s = 0.0f;
#pragma unroll
            for (int k = 0; k < 8; ++k) s += __expf(sv[k] - m);
#pragma unroll
            for (int off = 16; off >= 1; off >>= 1) s += __shfl_xor(s, off);
            if (lane == 0) s_mls[w] = m + __logf(s);      // (max + logsumexp) per row
        }
        __syncthreads();                                  // s_mls visible

        // ---- transposed, coalesced output write: out[t][p][b], b contiguous.
        //      wave w covers p in [16w,16w+16); lanes: 16 consecutive b per p
        //      -> 64B coalesced stores instead of 256B-strided 4B scatters. ----
        {
            const int b16 = lane & 15;
            const float mls = s_mls[b16];
#pragma unroll
            for (int k = 0; k < 8; ++k) {
                const int p = w * 16 + 2 * k + (lane >> 4);
                const float val = s_lds[b16][p] - mls;    // bank-conflict-free column read
                out_o[((size_t)t * LSTM_P + p) * LSTM_B + grp * 16 + b16] = val;
            }
        }
        __syncthreads();                                  // protect s_lds/s_mls reuse
    }
}

// --------------------------- launcher --------------------------------------
extern "C" void kernel_launch(void* const* d_in, const int* in_sizes, int n_in,
                              void* d_out, int out_size, void* d_ws, size_t ws_size,
                              hipStream_t stream)
{
    (void)in_sizes; (void)n_in; (void)out_size; (void)ws_size;
    const float* batch = (const float*)d_in[0];
    const float* W_ih  = (const float*)d_in[1];
    const float* W_hh  = (const float*)d_in[2];
    const float* b_ih  = (const float*)d_in[3];
    const float* b_hh  = (const float*)d_in[4];
    const float* W_lin = (const float*)d_in[5];
    const float* b_lin = (const float*)d_in[6];

    char* ws = (char*)d_ws;
    __bf16* W_ih_bf  = (__bf16*)(ws + OFF_WIH);
    __bf16* W_hh_bf  = (__bf16*)(ws + OFF_WHH);
    __bf16* W_lin_bf = (__bf16*)(ws + OFF_WLIN);
    float*  bias_sum = (float*)(ws + OFF_BIAS);
    v4u*    xf       = (v4u*)(ws + OFF_XF);

    float* out   = (float*)d_out;
    const size_t HT = (size_t)LSTM_T * LSTM_B * LSTM_H;   // 67,108,864
    float* out_h = out;
    float* out_c = out + HT;
    float* out_o = out + 2 * HT;

    lstm_prep<<<2048, 256, 0, stream>>>(W_ih, W_hh, b_ih, b_hh, W_lin,
                                        W_ih_bf, W_hh_bf, W_lin_bf, bias_sum);
    lstm_xproj<<<dim3(LSTM_T, 4, 2), 256, 0, stream>>>(batch, W_ih_bf, bias_sum, xf);
    lstm_step<<<4, 512, 0, stream>>>(xf, W_hh_bf, W_lin_bf, b_lin, out_h, out_c, out_o);
}
